// RWKV6_CrossAttention_Chunk_78391743086623
// MI455X (gfx1250) — matrix operand order repaired and checked by reading, not compiled, not run on hardware
//
#include <hip/hip_runtime.h>
#include <hip/hip_bf16.h>

// ---------------------------------------------------------------------------
// RWKV6-style cross-attention for MI455X (gfx1250, wave32, WMMA).
//  - All dense projections on v_wmma_f32_16x16x32_bf16 (bf16 in, f32 acc),
//    register-blocked 64x16 per wave (B-fragment reused by 4 WMMAs).
//  - Low-rank back-projections fuse the token-mix epilogue; decay fuses
//    exp(-exp(x)) into its GEMM epilogue.
//  - Sequential scan keeps the 64x64 state in registers per (b,head) and
//    double-buffers 16-timestep chunks of k/v/r/decay in LDS via
//    GLOBAL_LOAD_ASYNC_TO_LDS_B128 (ASYNCcnt pipeline).
// ---------------------------------------------------------------------------

typedef __attribute__((ext_vector_type(16))) __bf16 v16bf;
typedef __attribute__((ext_vector_type(8)))  float  v8f;

static constexpr int Bb  = 4;
static constexpr int Ll  = 2048;
static constexpr int Hh  = 256;      // hidden
static constexpr int NHq = 4;
static constexpr int BLq = Bb * Ll;  // 8192 tokens

// ---------------- helpers ----------------
__device__ __forceinline__ unsigned short f2bf(float f) {
  unsigned int u = __float_as_uint(f);
  unsigned int r = u + 0x7fffu + ((u >> 16) & 1u);   // round-to-nearest-even
  return (unsigned short)(r >> 16);
}

struct BfPack { uint4 lo, hi; };

__device__ __forceinline__ v16bf make_frag(const unsigned short* p0,
                                           const unsigned short* p1) {
  BfPack t;
  t.lo = *reinterpret_cast<const uint4*>(p0);
  t.hi = *reinterpret_cast<const uint4*>(p1);
  return __builtin_bit_cast(v16bf, t);
}

// async 16-byte global -> LDS copy, tracked by ASYNCcnt
__device__ __forceinline__ void async_ld16(const float* g, float* lds) {
  unsigned lo = (unsigned)(unsigned long long)(uintptr_t)lds; // LDS byte addr
  asm volatile("global_load_async_to_lds_b128 %0, %1, off"
               :: "v"(lo), "v"(g) : "memory");
}

// ---------------- elementwise kernels ----------------
__global__ void cvt_f32_bf16_kernel(unsigned short* __restrict__ dst,
                                    const float* __restrict__ src, int n) {
  int i = blockIdx.x * 256 + threadIdx.x;
  if (i < n) dst[i] = f2bf(src[i]);
}

// out = bf16(x + (x[l-1]-x)*mu[h]),  mu is a length-H vector
__global__ void mix_vec_kernel(unsigned short* __restrict__ out,
                               const float* __restrict__ keyval,
                               const float* __restrict__ mu) {
  int tok = blockIdx.x;           // b*L + l
  int h   = threadIdx.x;          // 0..255
  size_t idx = (size_t)tok * Hh + h;
  float x = keyval[idx];
  float prev = (tok % Ll == 0) ? 0.f : keyval[idx - Hh];
  out[idx] = f2bf(x + (prev - x) * mu[h]);
}

// ---------------- WMMA GEMM: C[M,N] = A[M,K] * W[N,K]^T ---------------------
// A,W bf16 row-major.  Each wave computes a 64x16 C slab (4 M-tiles sharing
// one B fragment per k-step).  Grids are exact multiples of 8 waves so EXEC
// is all-ones for every WMMA.
//
// Epilogues (EPI):
//   0: f32 store            (+bias if BIAS)
//   1: tanh -> bf16 store
//   2: exp(-exp(x)) -> f32  (decay)
//   3: token-mix -> bf16:   out = bf16(kv + (kv_prev - kv) * x)   (ldc==Hh)
template<int EPI, bool BIAS>
__global__ void gemm_bf16_kernel(const unsigned short* __restrict__ A, int lda,
                                 const unsigned short* __restrict__ W, int ldw,
                                 void* __restrict__ Cv, int ldc,
                                 const float* __restrict__ bias,
                                 const float* __restrict__ keyval,
                                 int Ntiles, int Kdim) {
  int wave = (blockIdx.x * blockDim.x + threadIdx.x) >> 5;
  int lane = threadIdx.x & 31;
  int tn = wave % Ntiles;          // N tile
  int tq = wave / Ntiles;          // M quad (4 tiles)
  int m0 = tq * 64, n0 = tn * 16;
  int row  = lane & 15;
  int half = lane >> 4;

  const unsigned short* Ap = A + (size_t)(m0 + row) * lda + half * 8;
  const unsigned short* Wp = W + (size_t)(n0 + row) * ldw + half * 8;
  const size_t astep = (size_t)16 * lda;

  v8f acc[4] = {v8f{}, v8f{}, v8f{}, v8f{}};
  for (int k = 0; k < Kdim; k += 32) {
    // speculative prefetch ahead on the streaming A operand
    __builtin_prefetch(Ap + k + 256, 0, 0);
    // B fragment: lanes 0-15 K {0-7,16-23}, lanes 16-31 K {8-15,24-31}
    v16bf bfrag = make_frag(Wp + k, Wp + k + 16);
#pragma unroll
    for (int s = 0; s < 4; ++s) {
      // A fragment: lanes 0-15 K {0-7,16-23}, lanes 16-31 K {8-15,24-31}
      v16bf a = make_frag(Ap + s * astep + k, Ap + s * astep + k + 16);
      acc[s] = __builtin_amdgcn_wmma_f32_16x16x32_bf16(
          false, a, false, bfrag, (short)0, acc[s], false, false);
    }
  }

  int n = n0 + row;
  float bv = BIAS ? bias[n] : 0.f;
#pragma unroll
  for (int s = 0; s < 4; ++s) {
#pragma unroll
    for (int e = 0; e < 8; ++e) {
      int m = m0 + s * 16 + half * 8 + e;
      float x = acc[s][e] + bv;
      if (EPI == 0) {
        reinterpret_cast<float*>(Cv)[(size_t)m * ldc + n] = x;
      } else if (EPI == 1) {
        reinterpret_cast<unsigned short*>(Cv)[(size_t)m * ldc + n] =
            f2bf(tanhf(x));
      } else if (EPI == 2) {
        reinterpret_cast<float*>(Cv)[(size_t)m * ldc + n] =
            __expf(-__expf(x));
      } else {  // EPI == 3 : token-mix with field mu = x
        size_t idx = (size_t)m * Hh + n;
        float kv   = keyval[idx];
        float prev = (m % Ll == 0) ? 0.f : keyval[idx - Hh];
        reinterpret_cast<unsigned short*>(Cv)[idx] =
            f2bf(kv + (prev - kv) * x);
      }
    }
  }
}

// ---------------- sequential linear-attention scan --------------------------
// One workgroup per (b,head).  State S[64x64] f32: thread t owns column
// col=t&63, rows d = (t>>6)*16 .. +15.  Chunks of 16 timesteps of k/v/r/decay
// are double-buffered in LDS via async global->LDS b128 copies: chunk c+1 is
// in flight while chunk c is consumed (ASYNCcnt returns in order, so waiting
// for <=4 outstanding guarantees the current chunk has landed).
static constexpr int CHq = 16;   // timesteps per chunk

__global__ void rec_kernel(float* __restrict__ o,
                           const float* __restrict__ r,
                           const float* __restrict__ k,
                           const float* __restrict__ v,
                           const float* __restrict__ w,
                           const float* __restrict__ u) {
  int b = blockIdx.x >> 2;         // NH = 4
  int h = blockIdx.x & 3;
  int t = threadIdx.x;
  int col = t & 63;
  int grp = t >> 6;

  // [buf][array: 0=k 1=v 2=r 3=w][step][elem]
  __shared__ float sbuf[2][4][CHq][64];
  __shared__ float red[2][4][64];

  float S[16];
  float ur[16];
#pragma unroll
  for (int i = 0; i < 16; ++i) {
    S[i]  = 0.f;
    ur[i] = u[h * 64 + grp * 16 + i];
  }

  const size_t headoff = (size_t)h * 64;
  const int lrow = t >> 4;         // 0..15  : timestep within chunk
  const int lq   = (t & 15) * 4;   // float4 offset within 64

  auto issue_chunk = [&](int cc, int pb) {
    size_t gbase = ((size_t)(b * Ll + cc * CHq + lrow)) * 256 + headoff + lq;
    async_ld16(&k[gbase], &sbuf[pb][0][lrow][lq]);
    async_ld16(&v[gbase], &sbuf[pb][1][lrow][lq]);
    async_ld16(&r[gbase], &sbuf[pb][2][lrow][lq]);
    async_ld16(&w[gbase], &sbuf[pb][3][lrow][lq]);
  };

  const int NC = Ll / CHq;
  issue_chunk(0, 0);               // prologue

  for (int c = 0; c < NC; ++c) {
    int cur = c & 1;
    if (c + 1 < NC) {
      issue_chunk(c + 1, cur ^ 1);                       // prefetch next chunk
      asm volatile("s_wait_asynccnt 0x4" ::: "memory");  // chunk c landed
    } else {
      asm volatile("s_wait_asynccnt 0x0" ::: "memory");
    }
    __syncthreads();   // LDS writes visible to all waves

    for (int s = 0; s < CHq; ++s) {
      float vcol = sbuf[cur][1][s][col];
      float p = 0.f, op = 0.f;
#pragma unroll
      for (int i = 0; i < 16; ++i) {
        int d = grp * 16 + i;
        float rd = sbuf[cur][2][s][d];
        float kd = sbuf[cur][0][s][d];
        p  += rd * ur[i] * kd;                        // r.(u*k) partial
        op += rd * S[i];                              // r.S[:,col] (pre-update)
        S[i] = sbuf[cur][3][s][d] * S[i] + kd * vcol; // state update
      }
      red[s & 1][grp][col] = op + p * vcol;
      __syncthreads();
      if (grp == 0) {
        size_t obase = ((size_t)(b * Ll + c * CHq + s)) * 256 + headoff + col;
        o[obase] = red[s & 1][0][col] + red[s & 1][1][col] +
                   red[s & 1][2][col] + red[s & 1][3][col];
      }
    }
  }
}

// ---------------- groupnorm (per head of 64) + SiLU gate --------------------
__global__ void gn_gate_kernel(unsigned short* __restrict__ out,
                               const float* __restrict__ o,
                               const float* __restrict__ g,
                               const float* __restrict__ gnw,
                               const float* __restrict__ gnb) {
  int tok = blockIdx.x;
  int t = threadIdx.x;          // 0..255 -> head = t>>6
  int head = t >> 6;

  __shared__ float buf[256];
  __shared__ float mean[4], rstd[4];

  size_t idx = (size_t)tok * 256 + t;
  float val = o[idx];
  buf[t] = val;
  __syncthreads();
  if ((t & 63) == 0) {
    float s = 0.f;
    for (int i = 0; i < 64; ++i) s += buf[head * 64 + i];
    mean[head] = s * (1.f / 64.f);
  }
  __syncthreads();
  float m = mean[head];
  buf[t] = (val - m) * (val - m);
  __syncthreads();
  if ((t & 63) == 0) {
    float s = 0.f;
    for (int i = 0; i < 64; ++i) s += buf[head * 64 + i];
    rstd[head] = rsqrtf(s * (1.f / 64.f) + 1e-5f);
  }
  __syncthreads();

  float normed = (val - m) * rstd[head] * gnw[t] + gnb[t];
  float gv = g[idx];
  float gate = gv / (1.f + __expf(-gv));   // g * sigmoid(g)
  out[idx] = f2bf(normed * gate);
}

// ---------------------------------------------------------------------------
extern "C" void kernel_launch(void* const* d_in, const int* in_sizes, int n_in,
                              void* d_out, int out_size, void* d_ws, size_t ws_size,
                              hipStream_t stream) {
  (void)in_sizes; (void)n_in; (void)out_size; (void)ws_size;

  const float* query  = (const float*)d_in[0];
  const float* keyval = (const float*)d_in[1];
  const float* kv0_mu = (const float*)d_in[2];
  const float* kv0_w  = (const float*)d_in[3];
  const float* kv2_w  = (const float*)d_in[4];
  const float* kv_bias= (const float*)d_in[5];
  const float* r_w    = (const float*)d_in[6];
  const float* k_w    = (const float*)d_in[7];
  const float* v_w    = (const float*)d_in[8];
  const float* g_w    = (const float*)d_in[9];
  const float* u      = (const float*)d_in[10];
  const float* w_mu   = (const float*)d_in[11];
  const float* wmu_a  = (const float*)d_in[12];
  const float* wmu_b  = (const float*)d_in[13];
  const float* wmu_bb = (const float*)d_in[14];
  const float* wp_a   = (const float*)d_in[15];
  const float* wp_b   = (const float*)d_in[16];
  const float* wp_bb  = (const float*)d_in[17];
  const float* gn_w   = (const float*)d_in[18];
  const float* gn_b   = (const float*)d_in[19];
  const float* o_w    = (const float*)d_in[20];

  // ---- carve workspace ----
  char* p = (char*)d_ws;
  auto take = [&](size_t bytes) -> char* {
    char* rp = p;
    p += (bytes + 255) & ~size_t(255);
    return rp;
  };
  auto takeu = [&](size_t elems) { return (unsigned short*)take(elems * 2); };
  auto takef = [&](size_t elems) { return (float*)take(elems * 4); };

  // bf16 weights
  unsigned short* wbf_kv0  = takeu(64 * 256);
  unsigned short* wbf_kv2  = takeu(256 * 64);
  unsigned short* wbf_r    = takeu(256 * 256);
  unsigned short* wbf_k    = takeu(256 * 256);
  unsigned short* wbf_v    = takeu(256 * 256);
  unsigned short* wbf_g    = takeu(256 * 256);
  unsigned short* wbf_wmua = takeu(32 * 256);
  unsigned short* wbf_wmub = takeu(256 * 32);
  unsigned short* wbf_wpa  = takeu(64 * 256);
  unsigned short* wbf_wpb  = takeu(256 * 64);
  unsigned short* wbf_o    = takeu(256 * 256);
  // bf16 activations
  unsigned short* q_bf   = takeu((size_t)BLq * 256);
  unsigned short* xm0_bf = takeu((size_t)BLq * 256);
  unsigned short* xw_bf  = takeu((size_t)BLq * 256);
  unsigned short* kv_bf  = takeu((size_t)BLq * 64);
  unsigned short* xk_bf  = takeu((size_t)BLq * 256);
  unsigned short* xv_bf  = takeu((size_t)BLq * 256);
  unsigned short* tmu_bf = takeu((size_t)BLq * 32);
  unsigned short* xp_bf  = takeu((size_t)BLq * 256);
  unsigned short* tp_bf  = takeu((size_t)BLq * 64);
  unsigned short* gat_bf = takeu((size_t)BLq * 256);
  // f32 intermediates
  float* dec_f = takef((size_t)BLq * 256);
  float* k_f   = takef((size_t)BLq * 256);
  float* v_f   = takef((size_t)BLq * 256);
  float* r_f   = takef((size_t)BLq * 256);
  float* g_f   = takef((size_t)BLq * 256);
  float* o_f   = takef((size_t)BLq * 256);

  auto cvt = [&](unsigned short* dst, const float* src, int n) {
    cvt_f32_bf16_kernel<<<(n + 255) / 256, 256, 0, stream>>>(dst, src, n);
  };

  // weight conversions (tiny)
  cvt(wbf_kv0, kv0_w, 64 * 256);
  cvt(wbf_kv2, kv2_w, 256 * 64);
  cvt(wbf_r, r_w, 256 * 256);
  cvt(wbf_k, k_w, 256 * 256);
  cvt(wbf_v, v_w, 256 * 256);
  cvt(wbf_g, g_w, 256 * 256);
  cvt(wbf_wmua, wmu_a, 32 * 256);
  cvt(wbf_wmub, wmu_b, 256 * 32);
  cvt(wbf_wpa, wp_a, 64 * 256);
  cvt(wbf_wpb, wp_b, 256 * 64);
  cvt(wbf_o, o_w, 256 * 256);
  cvt(q_bf, query, BLq * 256);

  // token mixes with vector mus
  mix_vec_kernel<<<BLq, 256, 0, stream>>>(xm0_bf, keyval, kv0_mu);
  mix_vec_kernel<<<BLq, 256, 0, stream>>>(xw_bf, keyval, w_mu);

  // grid: (Mquads=128) * Ntiles waves, 8 waves per block
  auto blocks = [](int Ntiles) { return (BLq / 64) * Ntiles / 8; };

  // kv = tanh(xm0 @ kv0_w^T) : N=64, K=256 -> bf16
  gemm_bf16_kernel<1, false><<<blocks(4), 256, 0, stream>>>(
      xm0_bf, 256, wbf_kv0, 256, kv_bf, 64, nullptr, nullptr, 4, 256);
  // xk = mix(keyval, kv[:, :32] @ kv2_w[:, :32]^T + bias0) : N=256, K=32
  gemm_bf16_kernel<3, true><<<blocks(16), 256, 0, stream>>>(
      kv_bf, 64, wbf_kv2, 64, xk_bf, 256, kv_bias, keyval, 16, 32);
  // xv = mix(keyval, kv[:, 32:] @ kv2_w[:, 32:]^T + bias1)
  gemm_bf16_kernel<3, true><<<blocks(16), 256, 0, stream>>>(
      kv_bf + 32, 64, wbf_kv2 + 32, 64, xv_bf, 256, kv_bias + 256, keyval,
      16, 32);

  // mu_dd path: t = tanh(xw @ wmu_a^T) (N=32), xp = mix(t @ wmu_b^T + bb)
  gemm_bf16_kernel<1, false><<<blocks(2), 256, 0, stream>>>(
      xw_bf, 256, wbf_wmua, 256, tmu_bf, 32, nullptr, nullptr, 2, 256);
  gemm_bf16_kernel<3, true><<<blocks(16), 256, 0, stream>>>(
      tmu_bf, 32, wbf_wmub, 32, xp_bf, 256, wmu_bb, keyval, 16, 32);

  // wlog path: tp = tanh(xp @ wp_a^T) (N=64), decay = exp(-exp(tp@wp_b^T+bb))
  gemm_bf16_kernel<1, false><<<blocks(4), 256, 0, stream>>>(
      xp_bf, 256, wbf_wpa, 256, tp_bf, 64, nullptr, nullptr, 4, 256);
  gemm_bf16_kernel<2, true><<<blocks(16), 256, 0, stream>>>(
      tp_bf, 64, wbf_wpb, 64, dec_f, 256, wp_bb, nullptr, 16, 64);

  // big projections: k, v, r, g  (N=256, K=256)
  gemm_bf16_kernel<0, false><<<blocks(16), 256, 0, stream>>>(
      xk_bf, 256, wbf_k, 256, k_f, 256, nullptr, nullptr, 16, 256);
  gemm_bf16_kernel<0, false><<<blocks(16), 256, 0, stream>>>(
      xv_bf, 256, wbf_v, 256, v_f, 256, nullptr, nullptr, 16, 256);
  gemm_bf16_kernel<0, false><<<blocks(16), 256, 0, stream>>>(
      q_bf, 256, wbf_r, 256, r_f, 256, nullptr, nullptr, 16, 256);
  gemm_bf16_kernel<0, false><<<blocks(16), 256, 0, stream>>>(
      q_bf, 256, wbf_g, 256, g_f, 256, nullptr, nullptr, 16, 256);

  // sequential scan, one workgroup per (b, head)
  rec_kernel<<<Bb * NHq, 256, 0, stream>>>(o_f, r_f, k_f, v_f, dec_f, u);

  // groupnorm + silu gate -> bf16
  gn_gate_kernel<<<BLq, 256, 0, stream>>>(gat_bf, o_f, g_f, gn_w, gn_b);

  // final projection straight into d_out (f32)
  gemm_bf16_kernel<0, false><<<blocks(16), 256, 0, stream>>>(
      gat_bf, 256, wbf_o, 256, (float*)d_out, 256, nullptr, nullptr, 16, 256);
}
